// RegiNet_CrossViTv3_SW_single3D_6262062318018
// MI455X (gfx1250) — compile-verified, matrix operand order checked by reading, not run on hardware
//
#include <hip/hip_runtime.h>
#include <cstdint>
#include <cstddef>

typedef __attribute__((ext_vector_type(16))) _Float16 v16h;
typedef __attribute__((ext_vector_type(8)))  float    v8f;

#define VOLN   128
#define VOL3   (128*128*128)
#define IMGN   128
#define RAYM   64
#define DIMN   192
#define TOKN   17
#define FFN_N  512

// ---------------------------------------------------------------- utilities
__device__ __forceinline__ float gelu_f(float x) {
    float x3 = x * x * x;
    float t  = tanhf(0.7978845608028654f * (x + 0.044715f * x3));
    return 0.5f * x * (1.0f + t);
}
__device__ __forceinline__ int clampi(int v, int lo, int hi) {
    return v < lo ? lo : (v > hi ? hi : v);
}

// K-index mapping for 16-bit A/B WMMA fragments (ISA 7.12.2)
__device__ __forceinline__ int wmma_klocal(int e, int half) {
    int v = e >> 1, p = e & 1;
    return ((v >> 2) << 4) + (half << 3) + ((v & 3) << 1) + p;
}

// ---------------------------------------------------------------- f32 -> f16
__global__ void cvt_f32_to_f16(const float* __restrict__ x, _Float16* __restrict__ o, long n) {
    long i = (long)blockIdx.x * blockDim.x + threadIdx.x;
    if (i < n) o[i] = (_Float16)x[i];
}

// ---------------------------------------------------------------- conv3d via implicit-GEMM WMMA
// in : (B, CIN, 128^3) f16      Wt : (COUT, CIN, 27) f32
// MODE 0: out f16 (B,COUT,vol), epilogue relu((conv+bias)*g/sqrt(1+eps)+be)
// MODE 1: out f32 (B,vol), epilogue conv+bias + x0 (residual), COUT==1
template <int CIN, int COUT, int MODE>
__global__ void __launch_bounds__(256)
conv3d_wmma(const _Float16* __restrict__ in, const float* __restrict__ Wt,
            const float* __restrict__ bias, const float* __restrict__ g,
            const float* __restrict__ be, const float* __restrict__ x0,
            void* __restrict__ outp, int nb) {
    const long tilesPerBatch = VOL3 / 16;
    int  wid  = threadIdx.x >> 5;
    int  lane = threadIdx.x & 31;
    long tile = (long)blockIdx.x * 8 + wid;
    long totalTiles = (long)nb * tilesPerBatch;
    if (tile >= totalTiles) return;                 // wave-uniform exit (EXEC stays all-1)
    int b     = (int)(tile / tilesPerBatch);
    int vbase = (int)((tile % tilesPerBatch) * 16);
    int half  = lane >> 4;
    int li    = lane & 15;
    int vox   = vbase + li;
    int vz = vox >> 14, vy = (vox >> 7) & 127, vx = vox & 127;

    const int KT  = CIN * 27;
    const int NKT = (KT + 31) / 32;
    v8f c = {};
    for (int kt = 0; kt < NKT; ++kt) {
        v16h a, bf;
#pragma unroll
        for (int e = 0; e < 16; ++e) {
            int k = kt * 32 + wmma_klocal(e, half);
            // A fragment: weights, row m = li (output channel)
            float aw = 0.0f;
            if (k < KT && li < COUT) {
                int cin = k / 27, t = k % 27;
                aw = Wt[(li * CIN + cin) * 27 + t];
            }
            a[e] = (_Float16)aw;
            // B fragment: im2col column n = li (voxel)
            _Float16 bv = (_Float16)0.0f;
            if (k < KT) {
                int cin = k / 27, t = k % 27;
                int dz = t / 9 - 1, dy = (t / 3) % 3 - 1, dx = t % 3 - 1;
                int zz = vz + dz, yy = vy + dy, xx = vx + dx;
                if ((unsigned)zz < 128u && (unsigned)yy < 128u && (unsigned)xx < 128u)
                    bv = in[((long)(b * CIN + cin) << 21) + (zz << 14) + (yy << 7) + xx];
            }
            bf[e] = bv;
        }
        c = __builtin_amdgcn_wmma_f32_16x16x32_f16(false, a, false, bf, (short)0, c, false, false);
    }
    if (MODE == 0) {
        _Float16* out = (_Float16*)outp;
        const float inv = rsqrtf(1.0f + 1e-5f);
#pragma unroll
        for (int r = 0; r < 8; ++r) {
            int m = r + (half << 3);
            if (m < COUT) {
                float vv = c[r] + bias[m];
                vv = vv * (g[m] * inv) + be[m];
                vv = fmaxf(vv, 0.0f);
                out[((long)(b * COUT + m) << 21) + vox] = (_Float16)vv;
            }
        }
    } else {
        float* out = (float*)outp;
        if (half == 0) {
            float vv = c[0] + bias[0] + x0[((long)b << 21) + vox];
            out[((long)b << 21) + vox] = vv;
        }
    }
}

// ---------------------------------------------------------------- generic WMMA GEMM
// C[b,m,n] = act(A[b,m,:] @ W[:,n] + bias[n]) + R[b,m,n]
__global__ void __launch_bounds__(128)
gemm_wmma(const float* __restrict__ A, long sA, int lda,
          const float* __restrict__ W, int ldw,
          const float* __restrict__ bias,
          const float* __restrict__ R, long sR, int ldr,
          float* __restrict__ C, long sC, int ldc,
          int M, int N, int K, int act) {
    int wid = threadIdx.x >> 5, lane = threadIdx.x & 31;
    int tm = (M + 15) >> 4, tn = (N + 15) >> 4;
    int tile = blockIdx.x * 4 + wid;
    if (tile >= tm * tn) return;                    // wave-uniform
    int mt = tile / tn, nt = tile % tn;
    int b  = blockIdx.z;
    int half = lane >> 4, li = lane & 15;
    int mrow = mt * 16 + li;
    int ncol = nt * 16 + li;
    v8f c = {};
    int nkt = (K + 31) >> 5;
    for (int kt = 0; kt < nkt; ++kt) {
        v16h a, w;
#pragma unroll
        for (int e = 0; e < 16; ++e) {
            int k = kt * 32 + wmma_klocal(e, half);
            float av = (k < K && mrow < M) ? A[b * sA + (long)mrow * lda + k] : 0.0f;
            a[e] = (_Float16)av;
            float wv = (k < K && ncol < N) ? W[(long)k * ldw + ncol] : 0.0f;
            w[e] = (_Float16)wv;
        }
        c = __builtin_amdgcn_wmma_f32_16x16x32_f16(false, a, false, w, (short)0, c, false, false);
    }
#pragma unroll
    for (int r = 0; r < 8; ++r) {
        int mm = mt * 16 + r + (half << 3);
        if (mm < M && ncol < N) {
            float v = c[r];
            if (bias) v += bias[ncol];
            if (act == 1) v = gelu_f(v);
            if (R) v += R[b * sR + (long)mm * ldr + ncol];
            C[b * sC + (long)mm * ldc + ncol] = v;
        }
    }
}

// ---------------------------------------------------------------- LayerNorm (one row per block)
__global__ void __launch_bounds__(256)
ln_kernel(const float* __restrict__ X, long sB, int ld,
          const float* __restrict__ g, const float* __restrict__ bta,
          float* __restrict__ Y, long sBy, int ldy, int D) {
    __shared__ float red[256];
    int row = blockIdx.x, b = blockIdx.y, t = threadIdx.x;
    const float* xr = X + b * sB + (long)row * ld;
    float v = (t < D) ? xr[t] : 0.0f;
    red[t] = v; __syncthreads();
    for (int s = 128; s > 0; s >>= 1) { if (t < s) red[t] += red[t + s]; __syncthreads(); }
    float mean = red[0] / D; __syncthreads();
    float d = (t < D) ? (v - mean) : 0.0f;
    red[t] = d * d; __syncthreads();
    for (int s = 128; s > 0; s >>= 1) { if (t < s) red[t] += red[t + s]; __syncthreads(); }
    float var = red[0] / D;
    if (t < D)
        Y[b * sBy + (long)row * ldy + t] = (v - mean) * rsqrtf(var + 1e-5f) * g[t] + bta[t];
}

// ---------------------------------------------------------------- softmax attention (small shapes)
__global__ void __launch_bounds__(128)
attn_kernel(const float* __restrict__ Q, long qsB, int qld, int qoff,
            const float* __restrict__ KV, long ksB, int kld, int koff, int voff,
            float* __restrict__ O, long osB, int oldd,
            int nb, int H, int HD, int Tq, int Tk) {
    int tid = blockIdx.x * blockDim.x + threadIdx.x;
    int total = nb * H * Tq;
    if (tid >= total) return;
    int qi = tid % Tq;
    int h  = (tid / Tq) % H;
    int b  = tid / (Tq * H);
    const float* q = Q + b * qsB + (long)qi * qld + qoff + h * HD;
    float sc[32];
    float mx = -1e30f;
    float scale = rsqrtf((float)HD);
    for (int kk = 0; kk < Tk; ++kk) {
        const float* kr = KV + b * ksB + (long)kk * kld + koff + h * HD;
        float s = 0.0f;
        for (int d = 0; d < HD; ++d) s += q[d] * kr[d];
        s *= scale; sc[kk] = s; mx = fmaxf(mx, s);
    }
    float sum = 0.0f;
    for (int kk = 0; kk < Tk; ++kk) { sc[kk] = __expf(sc[kk] - mx); sum += sc[kk]; }
    float inv = 1.0f / sum;
    for (int d = 0; d < HD; ++d) {
        float o = 0.0f;
        for (int kk = 0; kk < Tk; ++kk)
            o += sc[kk] * KV[b * ksB + (long)kk * kld + voff + h * HD + d];
        O[b * osB + (long)qi * oldd + h * HD + d] = o * inv;
    }
}

// ---------------------------------------------------------------- pose / dmin-dmax
__global__ void pose_kernel(const float* __restrict__ theta, const float* __restrict__ corner,
                            const float* __restrict__ param, float* __restrict__ mat,
                            float* __restrict__ dmm, int nb) {
    int b = blockIdx.x * blockDim.x + threadIdx.x;
    if (b >= nb) return;
    float r0 = theta[b * 6 + 0], r1 = theta[b * 6 + 1], r2 = theta[b * 6 + 2];
    float t0 = theta[b * 6 + 3], t1 = theta[b * 6 + 4], t2 = theta[b * 6 + 5];
    float ang = sqrtf(r0 * r0 + r1 * r1 + r2 * r2 + 1e-12f);
    float kx = r0 / ang, ky = r1 / ang, kz = r2 / ang;
    float s = sinf(ang), cc = cosf(ang);
    float K[9] = {0.f, -kz, ky, kz, 0.f, -kx, -ky, kx, 0.f};
    float K2[9];
    for (int i = 0; i < 3; ++i)
        for (int j = 0; j < 3; ++j) {
            float acc = 0.f;
            for (int k = 0; k < 3; ++k) acc += K[i * 3 + k] * K[k * 3 + j];
            K2[i * 3 + j] = acc;
        }
    float* m = mat + b * 12;
    for (int i = 0; i < 3; ++i)
        for (int j = 0; j < 3; ++j) {
            float iden = (i == j) ? 1.0f : 0.0f;
            m[i * 4 + j] = iden + s * K[i * 3 + j] + (1.0f - cc) * K2[i * 3 + j];
        }
    m[3] = t0; m[7] = t1; m[11] = t2;
    float src = param[0];
    float dmin = 1e30f, dmax = -1e30f;
    for (int n = 0; n < 8; ++n) {
        float cx = corner[b * 24 + n * 3 + 0];
        float cy = corner[b * 24 + n * 3 + 1];
        float cz = corner[b * 24 + n * 3 + 2];
        float p0 = m[0] * cx + m[1] * cy + m[2] * cz + m[3];
        float p1 = m[4] * cx + m[5] * cy + m[6] * cz + m[7];
        float p2 = m[8] * cx + m[9] * cy + m[10] * cz + m[11];
        float dz = p2 - src;
        float dist = sqrtf(p0 * p0 + p1 * p1 + dz * dz);
        dmin = fminf(dmin, dist); dmax = fmaxf(dmax, dist);
    }
    dmm[b * 2] = dmin; dmm[b * 2 + 1] = dmax;
}

// ---------------------------------------------------------------- ray-cast + trilinear projection
__global__ void __launch_bounds__(256)
project_kernel(const float* __restrict__ x3d, const float* __restrict__ mat,
               const float* __restrict__ dmm, const float* __restrict__ param,
               float* __restrict__ x2d, int nb) {
    int tid = blockIdx.x * blockDim.x + threadIdx.x;
    int total = nb * IMGN * IMGN;
    if (tid >= total) return;
    int j = tid % IMGN, i = (tid / IMGN) % IMGN, b = tid / (IMGN * IMGN);
    float src = param[0], det = param[1], pix = param[2];
    float dx = (j - (IMGN - 1) * 0.5f) * pix;     // xx
    float dy = (i - (IMGN - 1) * 0.5f) * pix;     // yy
    float dz = det - src;
    float inv = rsqrtf(dx * dx + dy * dy + dz * dz);
    dx *= inv; dy *= inv; dz *= inv;
    const float* m = mat + b * 12;
    float dmin = dmm[b * 2], dmax = dmm[b * 2 + 1];
    const float* v = x3d + ((long)b << 21);
    float acc = 0.0f;
    for (int mm = 0; mm < RAYM; ++mm) {
        float tt = dmin + (dmax - dmin) * (float)mm / (RAYM - 1);
        float px = dx * tt, py = dy * tt, pz = src + dz * tt;
        float g0 = m[0] * px + m[1] * py + m[2] * pz + m[3];
        float g1 = m[4] * px + m[5] * py + m[6] * pz + m[7];
        float g2 = m[8] * px + m[9] * py + m[10] * pz + m[11];
        float xf = (g0 + 1.0f) * 63.5f, yf = (g1 + 1.0f) * 63.5f, zf = (g2 + 1.0f) * 63.5f;
        float x0f = floorf(xf), y0f = floorf(yf), z0f = floorf(zf);
        float wx = xf - x0f, wy = yf - y0f, wz = zf - z0f;
        int x0 = clampi((int)x0f, 0, 127), x1 = clampi((int)x0f + 1, 0, 127);
        int y0 = clampi((int)y0f, 0, 127), y1 = clampi((int)y0f + 1, 0, 127);
        int z0 = clampi((int)z0f, 0, 127), z1 = clampi((int)z0f + 1, 0, 127);
        auto at = [&](int z, int y, int x) { return v[(z << 14) + (y << 7) + x]; };
        acc += at(z0, y0, x0) * (1 - wz) * (1 - wy) * (1 - wx)
             + at(z0, y0, x1) * (1 - wz) * (1 - wy) * wx
             + at(z0, y1, x0) * (1 - wz) * wy * (1 - wx)
             + at(z0, y1, x1) * (1 - wz) * wy * wx
             + at(z1, y0, x0) * wz * (1 - wy) * (1 - wx)
             + at(z1, y0, x1) * wz * (1 - wy) * wx
             + at(z1, y1, x0) * wz * wy * (1 - wx)
             + at(z1, y1, x1) * wz * wy * wx;
    }
    x2d[tid] = acc;
}

// ---------------------------------------------------------------- patchify (B,1,128,128) -> (B,16,1024)
__global__ void patch_kernel(const float* __restrict__ img, float* __restrict__ Pm, int nb) {
    int tid = blockIdx.x * blockDim.x + threadIdx.x;
    int total = nb * 16 * 1024;
    if (tid >= total) return;
    int e = tid & 1023, p = (tid >> 10) & 15, b = tid >> 14;
    int ph = p >> 2, pw = p & 3, a = e >> 5, bb = e & 31;
    Pm[tid] = img[b * (IMGN * IMGN) + (ph * 32 + a) * IMGN + pw * 32 + bb];
}

// ---------------------------------------------------------------- token assembly: cls/pos add
__global__ void tok_kernel(const float* __restrict__ emb, const float* __restrict__ cls,
                           const float* __restrict__ pos, float* __restrict__ tok, int nb) {
    int tid = blockIdx.x * blockDim.x + threadIdx.x;
    int total = nb * TOKN * DIMN;
    if (tid >= total) return;
    int d = tid % DIMN, t = (tid / DIMN) % TOKN, b = tid / (TOKN * DIMN);
    float base = (t == 0) ? cls[d] : emb[(b * 16 + (t - 1)) * DIMN + d];
    tok[tid] = base + pos[t * DIMN + d];
}

// ================================================================ host side
struct BranchW {
    const float *W1, *W2, *Wh, *Wo, *Wqkv, *b1, *b2, *bh, *bo, *bqkv, *cls,
                *hln_b, *hln_g, *ln1_b, *ln1_g, *ln2_b, *ln2_g, *pe_W, *pe_b, *pos;
};
struct CrossW {
    const float *Wkv, *Wo, *Wq, *bkv, *bo, *bq, *lnkv_b, *lnkv_g, *lnq_b, *lnq_g;
};
struct Scratch {
    float *xn, *qkv, *attno, *tok2, *ffn, *qn, *qproj, *kvbuf, *obuf;
};

static BranchW getBranch(void* const* d, int base) {
    BranchW p;
    p.W1    = (const float*)d[base + 0];  p.W2    = (const float*)d[base + 1];
    p.Wh    = (const float*)d[base + 2];  p.Wo    = (const float*)d[base + 3];
    p.Wqkv  = (const float*)d[base + 4];  p.b1    = (const float*)d[base + 5];
    p.b2    = (const float*)d[base + 6];  p.bh    = (const float*)d[base + 7];
    p.bo    = (const float*)d[base + 8];  p.bqkv  = (const float*)d[base + 9];
    p.cls   = (const float*)d[base + 10]; p.hln_b = (const float*)d[base + 11];
    p.hln_g = (const float*)d[base + 12]; p.ln1_b = (const float*)d[base + 13];
    p.ln1_g = (const float*)d[base + 14]; p.ln2_b = (const float*)d[base + 15];
    p.ln2_g = (const float*)d[base + 16]; p.pe_W  = (const float*)d[base + 17];
    p.pe_b  = (const float*)d[base + 18]; p.pos   = (const float*)d[base + 19];
    return p;
}
static CrossW getCross(void* const* d, int base) {
    CrossW p;
    p.Wkv    = (const float*)d[base + 0]; p.Wo     = (const float*)d[base + 1];
    p.Wq     = (const float*)d[base + 2]; p.bkv    = (const float*)d[base + 3];
    p.bo     = (const float*)d[base + 4]; p.bq     = (const float*)d[base + 5];
    p.lnkv_b = (const float*)d[base + 6]; p.lnkv_g = (const float*)d[base + 7];
    p.lnq_b  = (const float*)d[base + 8]; p.lnq_g  = (const float*)d[base + 9];
    return p;
}

static void launch_gemm(hipStream_t s, const float* A, long sA, int lda,
                        const float* W, int ldw, const float* bias,
                        const float* R, long sR, int ldr,
                        float* C, long sC, int ldc,
                        int M, int N, int K, int act, int batch) {
    int tm = (M + 15) / 16, tn = (N + 15) / 16;
    int tiles = tm * tn;
    dim3 grid((tiles + 3) / 4, 1, batch);
    gemm_wmma<<<grid, 128, 0, s>>>(A, sA, lda, W, ldw, bias, R, sR, ldr, C, sC, ldc, M, N, K, act);
}

static void encoder_layer(hipStream_t s, int nb, float* tok, int i, const BranchW& P, Scratch& S) {
    const long sTok = (long)TOKN * DIMN;
    // ln1
    ln_kernel<<<dim3(TOKN, nb), 256, 0, s>>>(tok, sTok, DIMN, P.ln1_g + i * DIMN, P.ln1_b + i * DIMN,
                                             S.xn, sTok, DIMN, DIMN);
    // qkv = xn @ Wqkv + bqkv
    launch_gemm(s, S.xn, sTok, DIMN, P.Wqkv + (long)i * DIMN * 3 * DIMN, 3 * DIMN,
                P.bqkv + i * 3 * DIMN, nullptr, 0, 0,
                S.qkv, (long)TOKN * 3 * DIMN, 3 * DIMN, TOKN, 3 * DIMN, DIMN, 0, nb);
    // self attention (h=4, hd=48)
    int total = nb * 4 * TOKN;
    attn_kernel<<<(total + 127) / 128, 128, 0, s>>>(S.qkv, (long)TOKN * 3 * DIMN, 3 * DIMN, 0,
                                                    S.qkv, (long)TOKN * 3 * DIMN, 3 * DIMN, DIMN, 2 * DIMN,
                                                    S.attno, sTok, DIMN, nb, 4, 48, TOKN, TOKN);
    // tok2 = tok + attno @ Wo + bo
    launch_gemm(s, S.attno, sTok, DIMN, P.Wo + (long)i * DIMN * DIMN, DIMN, P.bo + i * DIMN,
                tok, sTok, DIMN, S.tok2, sTok, DIMN, TOKN, DIMN, DIMN, 0, nb);
    // ln2
    ln_kernel<<<dim3(TOKN, nb), 256, 0, s>>>(S.tok2, sTok, DIMN, P.ln2_g + i * DIMN, P.ln2_b + i * DIMN,
                                             S.xn, sTok, DIMN, DIMN);
    // ffn = gelu(xn @ W1 + b1)
    launch_gemm(s, S.xn, sTok, DIMN, P.W1 + (long)i * DIMN * FFN_N, FFN_N, P.b1 + i * FFN_N,
                nullptr, 0, 0, S.ffn, (long)TOKN * FFN_N, FFN_N, TOKN, FFN_N, DIMN, 1, nb);
    // tok = tok2 + ffn @ W2 + b2
    launch_gemm(s, S.ffn, (long)TOKN * FFN_N, FFN_N, P.W2 + (long)i * FFN_N * DIMN, DIMN,
                P.b2 + i * DIMN, S.tok2, sTok, DIMN, tok, sTok, DIMN, TOKN, DIMN, FFN_N, 0, nb);
}

static void cross_layer(hipStream_t s, int nb, const float* tokQ, const float* tokKV,
                        float* tokDst, int i, const CrossW& P, Scratch& S) {
    const long sTok = (long)TOKN * DIMN;
    // qn = ln(cls of Q branch)  (row 0 only)
    ln_kernel<<<dim3(1, nb), 256, 0, s>>>(tokQ, sTok, DIMN, P.lnq_g + i * DIMN, P.lnq_b + i * DIMN,
                                          S.qn, DIMN, DIMN, DIMN);
    // kn = ln(tokKV)
    ln_kernel<<<dim3(TOKN, nb), 256, 0, s>>>(tokKV, sTok, DIMN, P.lnkv_g + i * DIMN, P.lnkv_b + i * DIMN,
                                             S.xn, sTok, DIMN, DIMN);
    // q = qn @ Wq + bq
    launch_gemm(s, S.qn, DIMN, DIMN, P.Wq + (long)i * DIMN * DIMN, DIMN, P.bq + i * DIMN,
                nullptr, 0, 0, S.qproj, DIMN, DIMN, 1, DIMN, DIMN, 0, nb);
    // kv = kn @ Wkv + bkv
    launch_gemm(s, S.xn, sTok, DIMN, P.Wkv + (long)i * DIMN * 2 * DIMN, 2 * DIMN, P.bkv + i * 2 * DIMN,
                nullptr, 0, 0, S.kvbuf, (long)TOKN * 2 * DIMN, 2 * DIMN, TOKN, 2 * DIMN, DIMN, 0, nb);
    // cross attention (h=8, hd=24, Tq=1)
    int total = nb * 8;
    attn_kernel<<<(total + 127) / 128, 128, 0, s>>>(S.qproj, DIMN, DIMN, 0,
                                                    S.kvbuf, (long)TOKN * 2 * DIMN, 2 * DIMN, 0, DIMN,
                                                    S.obuf, DIMN, DIMN, nb, 8, 24, 1, TOKN);
    // dst row0 = clsQ + o @ Wo + bo   (rest of dst already copied)
    launch_gemm(s, S.obuf, DIMN, DIMN, P.Wo + (long)i * DIMN * DIMN, DIMN, P.bo + i * DIMN,
                tokQ, sTok, DIMN, tokDst, sTok, DIMN, 1, DIMN, DIMN, 0, nb);
}

extern "C" void kernel_launch(void* const* d_in, const int* in_sizes, int n_in,
                              void* d_out, int out_size, void* d_ws, size_t ws_size,
                              hipStream_t stream) {
    // flat input indices (jax pytree order: sorted dict keys, recursively)
    const int IN_CORNER = 0, IN_PARAM = 1, IN_WOUT = 2, IN_BOUT = 3;
    const int CV = 4;                       // conv: W1..W5,b1..b5,be1..be4,g1..g4
    const int LG = 22, SM = 42, XLG = 62, XSM = 72;
    const int IN_THETA = 82, IN_X = 83;     // IN_Y = 84
    const int IN_Y = 84;

    const int nb = in_sizes[IN_X] / VOL3;

    const float* corner = (const float*)d_in[IN_CORNER];
    const float* param  = (const float*)d_in[IN_PARAM];
    const float* Wout   = (const float*)d_in[IN_WOUT];
    const float* bout   = (const float*)d_in[IN_BOUT];
    const float* cW[5]  = {(const float*)d_in[CV+0], (const float*)d_in[CV+1], (const float*)d_in[CV+2],
                           (const float*)d_in[CV+3], (const float*)d_in[CV+4]};
    const float* cb[5]  = {(const float*)d_in[CV+5], (const float*)d_in[CV+6], (const float*)d_in[CV+7],
                           (const float*)d_in[CV+8], (const float*)d_in[CV+9]};
    const float* cbe[4] = {(const float*)d_in[CV+10], (const float*)d_in[CV+11],
                           (const float*)d_in[CV+12], (const float*)d_in[CV+13]};
    const float* cg[4]  = {(const float*)d_in[CV+14], (const float*)d_in[CV+15],
                           (const float*)d_in[CV+16], (const float*)d_in[CV+17]};
    BranchW PL = getBranch(d_in, LG);
    BranchW PS = getBranch(d_in, SM);
    CrossW  XL = getCross(d_in, XLG);
    CrossW  XS = getCross(d_in, XSM);
    const float* theta = (const float*)d_in[IN_THETA];
    const float* x     = (const float*)d_in[IN_X];
    const float* y     = (const float*)d_in[IN_Y];

    // ---- workspace carving
    uint8_t* wp = (uint8_t*)d_ws;
    size_t off = 0;
    auto alloc = [&](size_t bytes) -> void* {
        void* p = wp + off;
        off = (off + bytes + 255) & ~(size_t)255;
        return p;
    };
    _Float16* actA = (_Float16*)alloc((size_t)nb * 8  * VOL3 * sizeof(_Float16));
    _Float16* actB = (_Float16*)alloc((size_t)nb * 16 * VOL3 * sizeof(_Float16));
    float* x3d  = (float*)alloc((size_t)nb * VOL3 * sizeof(float));
    float* matB = (float*)alloc((size_t)nb * 12 * sizeof(float));
    float* dmm  = (float*)alloc((size_t)nb * 2 * sizeof(float));
    float* x2d  = (float*)alloc((size_t)nb * IMGN * IMGN * sizeof(float));
    float* Pm   = (float*)alloc((size_t)nb * 16 * 1024 * sizeof(float));
    float* emb  = (float*)alloc((size_t)nb * 16 * DIMN * sizeof(float));
    float* tokS0 = (float*)alloc((size_t)nb * TOKN * DIMN * sizeof(float));
    float* tokS1 = (float*)alloc((size_t)nb * TOKN * DIMN * sizeof(float));
    float* tokL0 = (float*)alloc((size_t)nb * TOKN * DIMN * sizeof(float));
    float* tokL1 = (float*)alloc((size_t)nb * TOKN * DIMN * sizeof(float));
    Scratch S;
    S.xn    = (float*)alloc((size_t)nb * TOKN * DIMN * sizeof(float));
    S.qkv   = (float*)alloc((size_t)nb * TOKN * 3 * DIMN * sizeof(float));
    S.attno = (float*)alloc((size_t)nb * TOKN * DIMN * sizeof(float));
    S.tok2  = (float*)alloc((size_t)nb * TOKN * DIMN * sizeof(float));
    S.ffn   = (float*)alloc((size_t)nb * TOKN * FFN_N * sizeof(float));
    S.qn    = (float*)alloc((size_t)nb * DIMN * sizeof(float));
    S.qproj = (float*)alloc((size_t)nb * DIMN * sizeof(float));
    S.kvbuf = (float*)alloc((size_t)nb * TOKN * 2 * DIMN * sizeof(float));
    S.obuf  = (float*)alloc((size_t)nb * DIMN * sizeof(float));
    float* fsb  = (float*)alloc((size_t)nb * 256 * sizeof(float));
    float* feat = (float*)alloc((size_t)nb * 256 * sizeof(float));
    (void)ws_size; (void)n_in; (void)out_size;

    // ---- stage 1: 3D conv net (implicit-GEMM WMMA), x3d = x + convnet(x)
    {
        long nvox = (long)nb * VOL3;
        cvt_f32_to_f16<<<(int)((nvox + 255) / 256), 256, 0, stream>>>(x, actA, nvox);
        long tiles = (long)nb * (VOL3 / 16);
        int blocks = (int)((tiles + 7) / 8);
        conv3d_wmma<1, 4, 0><<<blocks, 256, 0, stream>>>(actA, cW[0], cb[0], cg[0], cbe[0], nullptr, actB, nb);
        conv3d_wmma<4, 8, 0><<<blocks, 256, 0, stream>>>(actB, cW[1], cb[1], cg[1], cbe[1], nullptr, actA, nb);
        conv3d_wmma<8, 16, 0><<<blocks, 256, 0, stream>>>(actA, cW[2], cb[2], cg[2], cbe[2], nullptr, actB, nb);
        conv3d_wmma<16, 8, 0><<<blocks, 256, 0, stream>>>(actB, cW[3], cb[3], cg[3], cbe[3], nullptr, actA, nb);
        conv3d_wmma<8, 1, 1><<<blocks, 256, 0, stream>>>(actA, cW[4], cb[4], nullptr, nullptr, x, x3d, nb);
    }

    // ---- stage 2: pose + ray-cast projection -> x2d
    pose_kernel<<<1, 32, 0, stream>>>(theta, corner, param, matB, dmm, nb);
    {
        int total = nb * IMGN * IMGN;
        project_kernel<<<(total + 255) / 256, 256, 0, stream>>>(x3d, matB, dmm, param, x2d, nb);
    }

    // ---- stage 3: patch embed + token assembly for both branches
    {
        int total = nb * 16 * 1024;
        const long sTok = (long)TOKN * DIMN;
        // sm branch from x2d
        patch_kernel<<<(total + 255) / 256, 256, 0, stream>>>(x2d, Pm, nb);
        launch_gemm(stream, Pm, 16L * 1024, 1024, PS.pe_W, DIMN, PS.pe_b, nullptr, 0, 0,
                    emb, 16L * DIMN, DIMN, 16, DIMN, 1024, 0, nb);
        int tt = nb * TOKN * DIMN;
        tok_kernel<<<(tt + 255) / 256, 256, 0, stream>>>(emb, PS.cls, PS.pos, tokS0, nb);
        // lg branch from y
        patch_kernel<<<(total + 255) / 256, 256, 0, stream>>>(y, Pm, nb);
        launch_gemm(stream, Pm, 16L * 1024, 1024, PL.pe_W, DIMN, PL.pe_b, nullptr, 0, 0,
                    emb, 16L * DIMN, DIMN, 16, DIMN, 1024, 0, nb);
        tok_kernel<<<(tt + 255) / 256, 256, 0, stream>>>(emb, PL.cls, PL.pos, tokL0, nb);
        (void)sTok;
    }

    // ---- stage 4: CrossViT trunk
    float *sm = tokS0, *smAlt = tokS1, *lg = tokL0, *lgAlt = tokL1;
    const size_t tokBytes = (size_t)nb * TOKN * DIMN * sizeof(float);
    for (int d = 0; d < 4; ++d) {
        for (int l = 0; l < 2; ++l) {
            int i = d * 2 + l;
            encoder_layer(stream, nb, sm, i, PS, S);
            encoder_layer(stream, nb, lg, i, PL, S);
        }
        for (int c = 0; c < 2; ++c) {
            int i = d * 2 + c;
            hipMemcpyAsync(smAlt, sm, tokBytes, hipMemcpyDeviceToDevice, stream);
            hipMemcpyAsync(lgAlt, lg, tokBytes, hipMemcpyDeviceToDevice, stream);
            cross_layer(stream, nb, sm, lg, smAlt, i, XS, S);   // sm2 = cross(sm, lg)
            cross_layer(stream, nb, lg, sm, lgAlt, i, XL, S);   // lg' = cross(lg, sm_old)
            float* t;
            t = sm; sm = smAlt; smAlt = t;
            t = lg; lg = lgAlt; lgAlt = t;
        }
    }

    // ---- stage 5: heads
    {
        const long sTok = (long)TOKN * DIMN;
        // fs = ln(sm[:,0]) @ Wh + bh
        ln_kernel<<<dim3(1, nb), 256, 0, stream>>>(sm, sTok, DIMN, PS.hln_g, PS.hln_b, S.qn, DIMN, DIMN, DIMN);
        launch_gemm(stream, S.qn, DIMN, DIMN, PS.Wh, 256, PS.bh, nullptr, 0, 0,
                    fsb, 256, 256, 1, 256, DIMN, 0, nb);
        // feat = fs + ln(lg[:,0]) @ Wh + bh
        ln_kernel<<<dim3(1, nb), 256, 0, stream>>>(lg, sTok, DIMN, PL.hln_g, PL.hln_b, S.qn, DIMN, DIMN, DIMN);
        launch_gemm(stream, S.qn, DIMN, DIMN, PL.Wh, 256, PL.bh, fsb, 256, 256,
                    feat, 256, 256, 1, 256, DIMN, 0, nb);
        // out = feat @ Wout + bout
        launch_gemm(stream, feat, 256, 256, Wout, 1, bout, nullptr, 0, 0,
                    (float*)d_out, 1, 1, 1, 1, 256, 0, nb);
    }
}
